// GPSDPAttention_78460462563681
// MI455X (gfx1250) — matrix operand order repaired and checked by reading, not compiled, not run on hardware
//
#include <hip/hip_runtime.h>

// ---------------------------------------------------------------------------
// Gaussian-prior causal attention for MI455X (gfx1250, wave32, WMMA).
//  outputs: context [B,H,S,D] fp32  ++  attn [B,H,S,S] fp32 (concatenated)
// Matmuls in f16 WMMA (16x16x32); softmax stats + outputs in fp32.
// v3: additive-mask gaussian table (no per-element predication/branches),
//     dual-accumulator WMMA chains (no hazard NOP bubbles).
// ---------------------------------------------------------------------------

typedef __attribute__((ext_vector_type(16))) _Float16 v16h;
typedef __attribute__((ext_vector_type(8)))  _Float16 v8h;
typedef __attribute__((ext_vector_type(8)))  float    v8f;

#define S_LEN   2048
#define DHEAD   128
#define NHEAD   16
#define NBATCH  2
#define QBLK    128      // query rows per workgroup (8 waves x 16)
#define JBLK    64       // key/value columns per LDS tile
#define KT_STR  72       // halves per row of transposed K tile (pad 64->72)
#define V_STR   136      // halves per row of V tile (pad 128->136)
#define P_STR   72       // halves per row of P tile
#define GPAD    128      // table covers dij in [-GPAD, S_LEN)
#define NEG_INF_F (-1.0e9f)

// contiguous 16-half load (16B-aligned by padding choices)
__device__ __forceinline__ v16h ld16h(const _Float16* p) {
  v8h lo = *(const v8h*)p;
  v8h hi = *(const v8h*)(p + 8);
  return __builtin_shufflevector(lo, hi, 0,1,2,3,4,5,6,7,8,9,10,11,12,13,14,15);
}

// A-matrix fragment per CDNA5 16-bit 16x32 layout:
//   lane m   (half==0): K = {d0..d0+7, d0+16..d0+23}
//   lane m+16(half==1): K = {d0+8..d0+15, d0+24..d0+31}
__device__ __forceinline__ v16h ldA(const _Float16* rowp, int d0, int half) {
  v8h lo = *(const v8h*)(rowp + d0 + half * 8);
  v8h hi = *(const v8h*)(rowp + d0 + 16 + half * 8);
  return __builtin_shufflevector(lo, hi, 0,1,2,3,4,5,6,7,8,9,10,11,12,13,14,15);
}

__device__ __forceinline__ v16h ldB(const _Float16* colp, int stride, int half) {
  v16h v;
  #pragma unroll
  for (int s = 0; s < 16; ++s)
    v[s] = colp[(8 * half + (s & 7) + 16 * (s >> 3)) * stride];
  return v;
}

__global__ __launch_bounds__(256)
void gpsdp_attn_kernel(const float* __restrict__ Qg,
                       const float* __restrict__ Kg,
                       const float* __restrict__ Vg,
                       float* __restrict__ ctx_out,
                       float* __restrict__ attn_out)
{
  // LDS layout (bytes):
  //   [0,      18432)  sKt  : Kt[DHEAD][KT_STR]   (transposed K, f16)
  //   [18432,  35840)  sV   : V[JBLK][V_STR]      (row-major V, f16)
  //   [35840,  54272)  sP   : P[QBLK][P_STR]      (probabilities, f16)
  //   [54272,  62976)  gtab : additive gaussian+mask table, 2176 fp32
  //                    gx[d] = exp(-d^2/2sg^2) (d>=0) | -1e9 (d<0)
  //   sQ (staging, 32KB) overlays [0, 32768)
  __shared__ __align__(16) unsigned char smem[62976];
  _Float16* sKt  = (_Float16*)(smem);
  _Float16* sV   = (_Float16*)(smem + 18432);
  _Float16* sP   = (_Float16*)(smem + 35840);
  float*    gtab = (float*)(smem + 54272);
  float*    gx   = gtab + GPAD;           // index by dij in [-128, 2047]
  _Float16* sQ   = (_Float16*)(smem);

  const int tid  = threadIdx.x;
  const int lane = tid & 31;
  const int wid  = tid >> 5;
  const int qb   = blockIdx.x;
  const int bh   = blockIdx.y;
  const int h    = bh & (NHEAD - 1);
  const int q0   = qb * QBLK;

  const size_t base = (size_t)bh * S_LEN * DHEAD;
  const float* Q = Qg + base;
  const float* K = Kg + base;
  const float* V = Vg + base;
  float* ctx  = ctx_out  + base;
  float* attn = attn_out + (size_t)bh * S_LEN * S_LEN;

  const float scale  = 0.0883883476483184f;            // 1/sqrt(128)
  const float sg     = (float)(h + 1);                 // sigma_h = h+1
  const float inv2s2 = 1.0f / (2.0f * sg * sg);

  // ---- additive gaussian/mask table ----
  for (int t = tid; t < S_LEN + GPAD; t += 256) {
    const int d = t - GPAD;
    gtab[t] = (d < 0) ? NEG_INF_F : __expf(-(float)(d * d) * inv2s2);
  }

  // ---- stage Q tile to LDS (f16, pre-scaled), build resident A-frags ----
  for (int idx = tid; idx < QBLK * DHEAD; idx += 256)
    sQ[idx] = (_Float16)(Q[(size_t)q0 * DHEAD + idx] * scale);
  __syncthreads();

  const int arow  = lane & 15;    // A-matrix row within 16-row tile
  const int ahalf = lane >> 4;    // which K-half this lane holds
  v16h qf[4];
  #pragma unroll
  for (int c = 0; c < 4; ++c)
    qf[c] = ldA(sQ + (wid * 16 + arow) * DHEAD, c * 32, ahalf);
  __syncthreads();                // before Kt overwrites the staging area

  const int njb = 2 * (qb + 1);   // causal: j-blocks 0 .. njb-1 cover j <= max i

  // C/D layout: element s[r] of lane L is row (r + 8*(L>>4)), col (L&15)
  const int rowi0 = q0 + wid * 16 + 8 * ahalf;

  float m[8], l[8];
  #pragma unroll
  for (int r = 0; r < 8; ++r) { m[r] = -3.0e38f; l[r] = 0.0f; }

  // =============== pass 1: online row max / sum ===============
  for (int jb = 0; jb < njb; ++jb) {
    const int k0 = jb * JBLK;
    if (jb + 1 < njb)   // prefetch next 32KB K tile (256 thr x 128B)
      __builtin_prefetch((const char*)(K + (size_t)(k0 + JBLK) * DHEAD) + tid * 128, 0, 1);
    for (int idx = tid; idx < JBLK * DHEAD; idx += 256) {
      int kk = idx >> 7, dd = idx & 127;
      sKt[dd * KT_STR + kk] = (_Float16)K[(size_t)(k0 + kk) * DHEAD + dd];
    }
    __syncthreads();

    // all four 16x16 score tiles of this j-block, kept in registers
    v8f sb[4];
    #pragma unroll
    for (int kt = 0; kt < 4; ++kt) {
      // two independent accumulator chains -> no WMMA->WMMA hazard stalls
      v8f s0 = {}, s1 = {};
      v16h bf0 = ldB(sKt + (0 * 32) * KT_STR + kt * 16 + (lane & 15), KT_STR, ahalf);
      v16h bf1 = ldB(sKt + (1 * 32) * KT_STR + kt * 16 + (lane & 15), KT_STR, ahalf);
      s0 = __builtin_amdgcn_wmma_f32_16x16x32_f16(false, qf[0], false, bf0,
                                                  (short)0, s0, false, false);
      s1 = __builtin_amdgcn_wmma_f32_16x16x32_f16(false, qf[1], false, bf1,
                                                  (short)0, s1, false, false);
      bf0 = ldB(sKt + (2 * 32) * KT_STR + kt * 16 + (lane & 15), KT_STR, ahalf);
      bf1 = ldB(sKt + (3 * 32) * KT_STR + kt * 16 + (lane & 15), KT_STR, ahalf);
      s0 = __builtin_amdgcn_wmma_f32_16x16x32_f16(false, qf[2], false, bf0,
                                                  (short)0, s0, false, false);
      s1 = __builtin_amdgcn_wmma_f32_16x16x32_f16(false, qf[3], false, bf1,
                                                  (short)0, s1, false, false);
      v8f s = s0 + s1;
      const int dij0 = rowi0 - (k0 + kt * 16 + (lane & 15));
      #pragma unroll
      for (int r = 0; r < 8; ++r)      // unconditional: table encodes the mask
        s[r] += gx[dij0 + r];
      sb[kt] = s;
    }

    // one lane-reduction per row per j-block
    #pragma unroll
    for (int r = 0; r < 8; ++r) {
      float mx = fmaxf(fmaxf(sb[0][r], sb[1][r]), fmaxf(sb[2][r], sb[3][r]));
      mx = fmaxf(mx, __shfl_xor(mx, 1, 16));
      mx = fmaxf(mx, __shfl_xor(mx, 2, 16));
      mx = fmaxf(mx, __shfl_xor(mx, 4, 16));
      mx = fmaxf(mx, __shfl_xor(mx, 8, 16));
      const float mn = fmaxf(m[r], mx);
      float ps = __expf(sb[0][r] - mn) + __expf(sb[1][r] - mn)
               + __expf(sb[2][r] - mn) + __expf(sb[3][r] - mn);
      ps += __shfl_xor(ps, 1, 16);
      ps += __shfl_xor(ps, 2, 16);
      ps += __shfl_xor(ps, 4, 16);
      ps += __shfl_xor(ps, 8, 16);
      l[r] = l[r] * __expf(m[r] - mn) + ps;
      m[r] = mn;
    }
    __syncthreads();
  }

  float invl[8];
  #pragma unroll
  for (int r = 0; r < 8; ++r) invl[r] = 1.0f / l[r];

  v8f acc[8];
  #pragma unroll
  for (int dt = 0; dt < 8; ++dt) acc[dt] = (v8f){};

  // ====== pass 2: recompute scores, emit attn, accumulate P x V ======
  for (int jb = 0; jb < njb; ++jb) {
    const int k0 = jb * JBLK;
    if (jb + 1 < njb) {  // prefetch next K and V tiles
      __builtin_prefetch((const char*)(K + (size_t)(k0 + JBLK) * DHEAD) + tid * 128, 0, 1);
      __builtin_prefetch((const char*)(V + (size_t)(k0 + JBLK) * DHEAD) + tid * 128, 0, 1);
    }
    for (int idx = tid; idx < JBLK * DHEAD; idx += 256) {
      int kk = idx >> 7, dd = idx & 127;
      sKt[dd * KT_STR + kk] = (_Float16)K[(size_t)(k0 + kk) * DHEAD + dd];
      sV[kk * V_STR + dd]   = (_Float16)V[(size_t)(k0 + kk) * DHEAD + dd];
    }
    __syncthreads();

    #pragma unroll
    for (int kt = 0; kt < 4; ++kt) {
      v8f s0 = {}, s1 = {};
      v16h bf0 = ldB(sKt + (0 * 32) * KT_STR + kt * 16 + (lane & 15), KT_STR, ahalf);
      v16h bf1 = ldB(sKt + (1 * 32) * KT_STR + kt * 16 + (lane & 15), KT_STR, ahalf);
      s0 = __builtin_amdgcn_wmma_f32_16x16x32_f16(false, qf[0], false, bf0,
                                                  (short)0, s0, false, false);
      s1 = __builtin_amdgcn_wmma_f32_16x16x32_f16(false, qf[1], false, bf1,
                                                  (short)0, s1, false, false);
      bf0 = ldB(sKt + (2 * 32) * KT_STR + kt * 16 + (lane & 15), KT_STR, ahalf);
      bf1 = ldB(sKt + (3 * 32) * KT_STR + kt * 16 + (lane & 15), KT_STR, ahalf);
      s0 = __builtin_amdgcn_wmma_f32_16x16x32_f16(false, qf[2], false, bf0,
                                                  (short)0, s0, false, false);
      s1 = __builtin_amdgcn_wmma_f32_16x16x32_f16(false, qf[3], false, bf1,
                                                  (short)0, s1, false, false);
      v8f s = s0 + s1;
      const int jcol = k0 + kt * 16 + (lane & 15);
      const int dij0 = rowi0 - jcol;
      #pragma unroll
      for (int r = 0; r < 8; ++r) {
        const float sc = s[r] + gx[dij0 + r];          // mask folded into table
        const float p  = __expf(sc - m[r]) * invl[r];  // masked -> 0
        attn[(size_t)(rowi0 + r) * S_LEN + jcol] = p;  // lanes 0..15: 64B lines
        sP[(wid * 16 + r + 8 * ahalf) * P_STR + kt * 16 + (lane & 15)] = (_Float16)p;
      }
    }
    __syncthreads();   // sP visible (and sV ready) for the PV matmul

    #pragma unroll
    for (int c = 0; c < 2; ++c) {                // two 32-wide K chunks of P
      v16h pf = ldA(sP + (wid * 16 + arow) * P_STR, c * 32, ahalf);
      #pragma unroll
      for (int dt = 0; dt < 8; ++dt) {           // 8 x 16-wide D tiles
        v16h vf = ldB(sV + (c * 32) * V_STR + dt * 16 + (lane & 15), V_STR, ahalf);
        acc[dt] = __builtin_amdgcn_wmma_f32_16x16x32_f16(false, pf, false, vf,
                                                         (short)0, acc[dt],
                                                         false, false);
      }
    }
    __syncthreads();   // before next iteration overwrites Kt/V/P
  }

  // ---- write context ----
  #pragma unroll
  for (int dt = 0; dt < 8; ++dt) {
    #pragma unroll
    for (int r = 0; r < 8; ++r) {
      const int i = rowi0 + r;
      ctx[(size_t)i * DHEAD + dt * 16 + (lane & 15)] = acc[dt][r];
    }
  }

  // ---- zero the fully-masked attn columns (j >= njb*JBLK) ----
  const int cstart = njb * JBLK;
  if (cstart < S_LEN) {
    const int zc  = S_LEN - cstart;       // multiple of 128
    const int zh  = tid & 1;              // 2 threads per row
    const int row = tid >> 1;             // 0..127
    float4* p4 = (float4*)(attn + (size_t)(q0 + row) * S_LEN + cstart
                           + (size_t)zh * (zc >> 1));
    const int n4 = zc >> 3;               // (zc/2)/4 float4 stores
    for (int c = 0; c < n4; ++c) p4[c] = make_float4(0.f, 0.f, 0.f, 0.f);
  }
}

extern "C" void kernel_launch(void* const* d_in, const int* in_sizes, int n_in,
                              void* d_out, int out_size, void* d_ws, size_t ws_size,
                              hipStream_t stream) {
  (void)in_sizes; (void)n_in; (void)out_size; (void)d_ws; (void)ws_size;
  const float* Q = (const float*)d_in[0];
  const float* K = (const float*)d_in[1];
  const float* V = (const float*)d_in[2];
  // d_in[3] is the causal mask; causality is hard-coded in the kernel.
  float* ctx  = (float*)d_out;
  float* attn = (float*)d_out + (size_t)NBATCH * NHEAD * S_LEN * DHEAD;

  dim3 grid(S_LEN / QBLK, NBATCH * NHEAD);
  gpsdp_attn_kernel<<<grid, 256, 0, stream>>>(Q, K, V, ctx, attn);
}
